// AutoregressiveGroupQuerySelfAttention_14534169330028
// MI455X (gfx1250) — matrix operand order repaired and checked by reading, not compiled, not run on hardware
//
#include <hip/hip_runtime.h>
#include <hip/hip_bf16.h>
#include <stdint.h>

// ---------------------------------------------------------------------------
// MI455X (gfx1250) bf16-WMMA causal multi-head attention.
// GEMM tiles staged with the Tensor Data Mover (TENSORcnt); attention tiles
// staged with async global->LDS (ASYNCcnt); all matmuls on
// v_wmma_f32_16x16x32_bf16 with f32 accumulation.
// ---------------------------------------------------------------------------

typedef __attribute__((ext_vector_type(16))) __bf16 v16bf;
typedef __attribute__((ext_vector_type(8)))  __bf16 v8bf;
typedef __attribute__((ext_vector_type(8)))  float  v8f;
typedef __attribute__((ext_vector_type(4)))  unsigned v4u;
typedef __attribute__((ext_vector_type(8)))  int      v8i;
typedef __attribute__((ext_vector_type(4)))  int      v4i;

#define NH   16
#define HD   128
#define SEQ  2048
#define HID  2048
#define BATCH 2

#if __has_builtin(__builtin_amdgcn_tensor_load_to_lds)
#define USE_TDM 1
#else
#define USE_TDM 0
#endif

__device__ __forceinline__ __bf16 f2bf(float f) {
    unsigned u = __builtin_bit_cast(unsigned, f);
    unsigned r = (u + 0x7FFFu + ((u >> 16) & 1u)) >> 16;
    unsigned short s = (unsigned short)r;
    return __builtin_bit_cast(__bf16, s);
}
__device__ __forceinline__ float bf2f(__bf16 b) {
    unsigned short s = __builtin_bit_cast(unsigned short, b);
    unsigned u = ((unsigned)s) << 16;
    return __builtin_bit_cast(float, u);
}

// ----- async global->LDS (ASYNCcnt) ----------------------------------------
__device__ __forceinline__ void async_b128(const __bf16* g, __bf16* l) {
    unsigned lds = (unsigned)(uintptr_t)l;
    unsigned long long ga = (unsigned long long)(uintptr_t)g;
    asm volatile("global_load_async_to_lds_b128 %0, %1, off"
                 :: "v"(lds), "v"(ga) : "memory");
}
__device__ __forceinline__ void wait_async0() {
#if __has_builtin(__builtin_amdgcn_s_wait_asynccnt)
    __builtin_amdgcn_s_wait_asynccnt(0);
#else
    asm volatile("s_wait_asynccnt 0x0" ::: "memory");
#endif
}

// ----- Tensor Data Mover (TENSORcnt) ---------------------------------------
// 2D tile load: tile_h x tile_w bf16 elements from a row-major tensor with
// row stride `row_stride` (elements) into a packed LDS tile.  D# layout per
// CDNA5 ISA ch.8 (group0: count/lds/global/type, group1: dims/strides).
__device__ __forceinline__ void tdm_load_2d(const __bf16* gsrc, unsigned lds_off,
                                            unsigned tensor_w, unsigned tensor_h,
                                            unsigned tile_w, unsigned tile_h,
                                            unsigned row_stride) {
#if USE_TDM
    unsigned long long ga = (unsigned long long)(uintptr_t)gsrc;
    v4u g0;
    g0.x = 1u;                                        // count=1, user D#
    g0.y = lds_off;                                   // lds_addr [63:32]
    g0.z = (unsigned)ga;                              // global_addr[31:0]
    g0.w = (unsigned)(ga >> 32) | 0x80000000u;        // ga[56:32] | type=2
    v8i g1;
    g1[0] = (int)(1u << 16);                          // data_size=1 (2 bytes)
    g1[1] = (int)((tensor_w & 0xFFFFu) << 16);        // tensor_dim0[15:0]
    g1[2] = (int)((tensor_w >> 16) | ((tensor_h & 0xFFFFu) << 16));
    g1[3] = (int)((tensor_h >> 16) | (tile_w << 16)); // tile_dim0
    g1[4] = (int)tile_h;                              // tile_dim1 (dim2=0)
    g1[5] = (int)row_stride;                          // dim0_stride[31:0]
    g1[6] = 0;
    g1[7] = 0;
    v4i z4 = {0, 0, 0, 0};
#if __clang_major__ >= 23
    v8i z8 = {0, 0, 0, 0, 0, 0, 0, 0};
    __builtin_amdgcn_tensor_load_to_lds(g0, g1, z4, z4, z8, 0);
#else
    __builtin_amdgcn_tensor_load_to_lds(g0, g1, z4, z4, 0);
#endif
#else
    (void)gsrc; (void)lds_off; (void)tensor_w; (void)tensor_h;
    (void)tile_w; (void)tile_h; (void)row_stride;
#endif
}
__device__ __forceinline__ void wait_tensor0() {
#if __has_builtin(__builtin_amdgcn_s_wait_tensorcnt)
    __builtin_amdgcn_s_wait_tensorcnt(0);
#else
    asm volatile("s_wait_tensorcnt 0x0" ::: "memory");
#endif
}

// ----- WMMA fragments (CDNA5 ISA 7.12.2 layouts) ---------------------------
__device__ __forceinline__ v16bf load_a_frag(const __bf16* p, int ld, int row, int k0) {
    int lane = threadIdx.x & 31;
    int half = lane >> 4;
    int m    = lane & 15;
    const __bf16* r = p + (size_t)(row + m) * ld + k0 + half * 8;
    v16bf out;
    *((v8bf*)&out)       = *(const v8bf*)(r);
    *(((v8bf*)&out) + 1) = *(const v8bf*)(r + 16);
    return out;
}
__device__ __forceinline__ v16bf load_b_frag(const __bf16* p, int ld, int col, int k0) {
    int lane = threadIdx.x & 31;
    int half = lane >> 4;
    int n    = lane & 15;
    const __bf16* r = p + (size_t)(col + n) * ld + k0 + half * 8;
    v16bf out;
    *((v8bf*)&out)       = *(const v8bf*)(r);
    *(((v8bf*)&out) + 1) = *(const v8bf*)(r + 16);
    return out;
}
__device__ __forceinline__ v8f wmma_bf16(v16bf a, v16bf b, v8f c) {
    return __builtin_amdgcn_wmma_f32_16x16x32_bf16(
        false, a, false, b, (short)0, c, false, false);
}

// ---------------------------------------------------------------------------
// Kernel 1: fp32 -> bf16 conversion (grid-stride).
// ---------------------------------------------------------------------------
__global__ void convert_bf16_kernel(const float* __restrict__ in,
                                    __bf16* __restrict__ out, long long n) {
    long long stride = (long long)gridDim.x * blockDim.x;
    for (long long i = (long long)blockIdx.x * blockDim.x + threadIdx.x; i < n; i += stride)
        out[i] = f2bf(in[i]);
}

// ---------------------------------------------------------------------------
// Kernel 2: fused QKV NT-GEMM.  C[s,o] = sum_h x[s,h] * W[o,h].
// 128x128 tile/block, 8 waves (2x4), each wave 64x32 via 4x2 wmma accs.
// A/B K-tiles (128x32 bf16 = 8KB) staged by the TDM (one tensor_load_to_lds
// per tile, issued by wave 0 only -- TDM ignores EXEC so the issue is gated
// by a scalar branch), double buffered on TENSORcnt + block barrier.
// ---------------------------------------------------------------------------
__global__ __launch_bounds__(256) void qkv_gemm_kernel(
        const __bf16* __restrict__ xbf,
        const __bf16* __restrict__ wq, const __bf16* __restrict__ wk,
        const __bf16* __restrict__ wv,
        __bf16* __restrict__ Qbf, __bf16* __restrict__ Kbf,
        __bf16* __restrict__ Vt) {
    __shared__ __align__(16) __bf16 lA[2][128][32];
    __shared__ __align__(16) __bf16 lB[2][128][32];

    const int which = blockIdx.z;
    const __bf16* W = (which == 0) ? wq : (which == 1) ? wk : wv;
    const int Nbase = blockIdx.x * 128;
    const int Mbase = blockIdx.y * 128;
    const int wave = threadIdx.x >> 5;
    const int Mwl = (wave >> 2) * 64;
    const int Nwl = (wave & 3) * 32;

#if USE_TDM
    const bool issuer = ((__builtin_amdgcn_readfirstlane(threadIdx.x) >> 5) == 0);
#else
    const int trow = threadIdx.x >> 1;
    const int tcol = (threadIdx.x & 1) * 16;
#endif

    auto prefetch = [&](int k0, int buf) {
#if USE_TDM
        if (issuer) {
            tdm_load_2d(xbf + (size_t)Mbase * HID + k0,
                        (unsigned)(uintptr_t)&lA[buf][0][0],
                        HID, 1u << 20, 32, 128, HID);
            tdm_load_2d(W + (size_t)Nbase * HID + k0,
                        (unsigned)(uintptr_t)&lB[buf][0][0],
                        HID, 1u << 20, 32, 128, HID);
        }
#else
        const __bf16* ga = xbf + (size_t)(Mbase + trow) * HID + k0 + tcol;
        async_b128(ga,     &lA[buf][trow][tcol]);
        async_b128(ga + 8, &lA[buf][trow][tcol + 8]);
        const __bf16* gb = W + (size_t)(Nbase + trow) * HID + k0 + tcol;
        async_b128(gb,     &lB[buf][trow][tcol]);
        async_b128(gb + 8, &lB[buf][trow][tcol + 8]);
#endif
    };
    auto wait_tiles = [&]() {
#if USE_TDM
        wait_tensor0();      // no-op for non-issuer waves (TENSORcnt==0)
#else
        wait_async0();
#endif
    };

    v8f acc[4][2] = {};
    prefetch(0, 0);
    for (int k0 = 0; k0 < HID; k0 += 32) {
        const int buf = (k0 >> 5) & 1;
        wait_tiles();
        __syncthreads();
        if (k0 + 32 < HID) prefetch(k0 + 32, buf ^ 1);

        v16bf a[4], b[2];
        #pragma unroll
        for (int i = 0; i < 4; ++i) a[i] = load_a_frag(&lA[buf][0][0], 32, Mwl + i * 16, 0);
        #pragma unroll
        for (int j = 0; j < 2; ++j) b[j] = load_b_frag(&lB[buf][0][0], 32, Nwl + j * 16, 0);
        #pragma unroll
        for (int i = 0; i < 4; ++i)
            #pragma unroll
            for (int j = 0; j < 2; ++j)
                acc[i][j] = wmma_bf16(a[i], b[j], acc[i][j]);
    }

    const int lane = threadIdx.x & 31, half = lane >> 4, nl = lane & 15;
    #pragma unroll
    for (int i = 0; i < 4; ++i)
        #pragma unroll
        for (int j = 0; j < 2; ++j) {
            const int col = Nbase + Nwl + j * 16 + nl;
            const int head = col >> 7, d = col & (HD - 1);
            #pragma unroll
            for (int r = 0; r < 8; ++r) {
                const int rowg = Mbase + Mwl + i * 16 + r + 8 * half;   // 0..B*S-1
                const int bb = rowg >> 11, s = rowg & (SEQ - 1);
                const __bf16 bv = f2bf(acc[i][j][r]);
                if (which == 2)
                    Vt[(((size_t)bb * NH + head) * HD + d) * SEQ + s] = bv;
                else {
                    __bf16* O = (which == 0) ? Qbf : Kbf;
                    O[(((size_t)bb * NH + head) * SEQ + s) * HD + d] = bv;
                }
            }
        }
}

// ---------------------------------------------------------------------------
// Kernel 3: in-place RoPE on Q and K ([b,h,s,d] bf16).
// Folds the reference's score scale (*sqrt(D)) into Q.
// ---------------------------------------------------------------------------
__global__ void rope_kernel(__bf16* __restrict__ Q, __bf16* __restrict__ K) {
    const long long N = (long long)BATCH * NH * SEQ * (HD / 2);
    const long long stride = (long long)gridDim.x * blockDim.x;
    for (long long idx = (long long)blockIdx.x * blockDim.x + threadIdx.x;
         idx < 2 * N; idx += stride) {
        const int isK = idx >= N;
        const long long t = isK ? idx - N : idx;
        const int i = (int)(t & 63);
        const long long rest = t >> 6;
        const int s = (int)(rest & (SEQ - 1));
        const long long bh = rest >> 11;
        __bf16* P = (isK ? K : Q) + (bh * SEQ + s) * HD;
        const float ang = (float)s * __powf(10000.0f, -(float)i / 64.0f);
        float c, sn;
        __sincosf(ang, &sn, &c);
        const float scale = isK ? 1.0f : 11.313708499f; // sqrt(128)
        const float x1 = bf2f(P[i]), x2 = bf2f(P[i + 64]);
        P[i]      = f2bf((x1 * c - x2 * sn) * scale);
        P[i + 64] = f2bf((x2 * c + x1 * sn) * scale);
    }
}

// ---------------------------------------------------------------------------
// Kernel 4: flash-attention, causal.  Grid = (S/128 qtiles, NH, B),
// 256 threads = 8 waves; each wave owns 16 query rows.
// K/V tiles staged once per block via async global->LDS (double buffered).
// P re-laid out C-frag -> A-frag through per-wave LDS (DScnt-ordered).
// ---------------------------------------------------------------------------
__global__ __launch_bounds__(256) void attn_kernel(
        const __bf16* __restrict__ Q, const __bf16* __restrict__ K,
        const __bf16* __restrict__ Vt, __bf16* __restrict__ ctx) {
    __shared__ __align__(16) __bf16 kT[2][32][128];   // [key][d]
    __shared__ __align__(16) __bf16 vT[2][128][32];   // [d][key]
    __shared__ __align__(16) __bf16 p_lds[8][16][32];

    const int qtile = blockIdx.x, head = blockIdx.y, b = blockIdx.z;
    const int wave = threadIdx.x >> 5;
    const int lane = threadIdx.x & 31, halfl = lane >> 4, nl = lane & 15;

    const __bf16* Qh = Q + ((size_t)b * NH + head) * SEQ * HD;
    const __bf16* Kh = K + ((size_t)b * NH + head) * SEQ * HD;
    const __bf16* Vh = Vt + ((size_t)b * NH + head) * HD * SEQ;

    const int q0 = qtile * 128 + wave * 16;

    const int krow = threadIdx.x >> 3, kcol = (threadIdx.x & 7) * 16;   // 32x128
    const int vrow = threadIdx.x >> 1, vcol = (threadIdx.x & 1) * 16;   // 128x32

    auto prefetch = [&](int kb, int buf) {
        const __bf16* gk = Kh + (size_t)(kb + krow) * HD + kcol;
        async_b128(gk,     &kT[buf][krow][kcol]);
        async_b128(gk + 8, &kT[buf][krow][kcol + 8]);
        const __bf16* gv = Vh + (size_t)vrow * SEQ + kb + vcol;
        async_b128(gv,     &vT[buf][vrow][vcol]);
        async_b128(gv + 8, &vT[buf][vrow][vcol + 8]);
    };

    v16bf qf[4];
    #pragma unroll
    for (int t = 0; t < 4; ++t) qf[t] = load_a_frag(Qh, HD, q0, t * 32);

    v8f oacc[8] = {};
    float row_max[8], row_sum[8];
    #pragma unroll
    for (int r = 0; r < 8; ++r) { row_max[r] = -3.0e38f; row_sum[r] = 0.0f; }

    const int kend = qtile * 128 + 128;           // uniform across the block
    prefetch(0, 0);
    for (int kb = 0; kb < kend; kb += 32) {
        const int buf = (kb >> 5) & 1;
        wait_async0();
        __syncthreads();
        if (kb + 32 < kend) prefetch(kb + 32, buf ^ 1);

        v8f s0 = {}, s1 = {};
        #pragma unroll
        for (int t = 0; t < 4; ++t) {
            v16bf b0 = load_b_frag(&kT[buf][0][0], HD, 0, t * 32);
            v16bf b1 = load_b_frag(&kT[buf][0][0], HD, 16, t * 32);
            s0 = wmma_bf16(qf[t], b0, s0);
            s1 = wmma_bf16(qf[t], b1, s1);
        }
        #pragma unroll
        for (int r = 0; r < 8; ++r) {
            const int qrow = q0 + r + 8 * halfl;
            const int k0g = kb + nl, k1g = kb + 16 + nl;
            const float v0 = (k0g <= qrow) ? s0[r] : -3.0e38f;
            const float v1 = (k1g <= qrow) ? s1[r] : -3.0e38f;
            float mx = fmaxf(v0, v1);
            #pragma unroll
            for (int m = 1; m < 16; m <<= 1) mx = fmaxf(mx, __shfl_xor(mx, m, 32));
            const float nm = fmaxf(row_max[r], mx);
            const float corr = __expf(row_max[r] - nm);
            const float p0 = (k0g <= qrow) ? __expf(s0[r] - nm) : 0.0f;
            const float p1 = (k1g <= qrow) ? __expf(s1[r] - nm) : 0.0f;
            float ps = p0 + p1;
            #pragma unroll
            for (int m = 1; m < 16; m <<= 1) ps += __shfl_xor(ps, m, 32);
            row_sum[r] = row_sum[r] * corr + ps;
            row_max[r] = nm;
            #pragma unroll
            for (int t = 0; t < 8; ++t) oacc[t][r] *= corr;
            p_lds[wave][r + 8 * halfl][nl]      = f2bf(p0);
            p_lds[wave][r + 8 * halfl][16 + nl] = f2bf(p1);
        }
        asm volatile("s_wait_dscnt 0x0" ::: "memory");
        v16bf pf = load_a_frag(&p_lds[wave][0][0], 32, 0, 0);
        #pragma unroll
        for (int t = 0; t < 8; ++t) {
            v16bf vf = load_b_frag(&vT[buf][0][0], 32, t * 16, 0);
            oacc[t] = wmma_bf16(pf, vf, oacc[t]);
        }
    }

    #pragma unroll
    for (int t = 0; t < 8; ++t) {
        const int d = t * 16 + nl;
        #pragma unroll
        for (int r = 0; r < 8; ++r) {
            const int qrow = q0 + r + 8 * halfl;
            const float val = oacc[t][r] / row_sum[r];
            ctx[((size_t)b * SEQ + qrow) * HID + head * HD + d] = f2bf(val);
        }
    }
}

// ---------------------------------------------------------------------------
// Kernel 5: output projection NT-GEMM (TDM-staged), fp32 result.
// ---------------------------------------------------------------------------
__global__ __launch_bounds__(256) void out_gemm_kernel(
        const __bf16* __restrict__ ctx, const __bf16* __restrict__ wo,
        float* __restrict__ out) {
    __shared__ __align__(16) __bf16 lA[2][128][32];
    __shared__ __align__(16) __bf16 lB[2][128][32];

    const int Nbase = blockIdx.x * 128;
    const int Mbase = blockIdx.y * 128;
    const int wave = threadIdx.x >> 5;
    const int Mwl = (wave >> 2) * 64;
    const int Nwl = (wave & 3) * 32;

#if USE_TDM
    const bool issuer = ((__builtin_amdgcn_readfirstlane(threadIdx.x) >> 5) == 0);
#else
    const int trow = threadIdx.x >> 1;
    const int tcol = (threadIdx.x & 1) * 16;
#endif

    auto prefetch = [&](int k0, int buf) {
#if USE_TDM
        if (issuer) {
            tdm_load_2d(ctx + (size_t)Mbase * HID + k0,
                        (unsigned)(uintptr_t)&lA[buf][0][0],
                        HID, 1u << 20, 32, 128, HID);
            tdm_load_2d(wo + (size_t)Nbase * HID + k0,
                        (unsigned)(uintptr_t)&lB[buf][0][0],
                        HID, 1u << 20, 32, 128, HID);
        }
#else
        const __bf16* ga = ctx + (size_t)(Mbase + trow) * HID + k0 + tcol;
        async_b128(ga,     &lA[buf][trow][tcol]);
        async_b128(ga + 8, &lA[buf][trow][tcol + 8]);
        const __bf16* gb = wo + (size_t)(Nbase + trow) * HID + k0 + tcol;
        async_b128(gb,     &lB[buf][trow][tcol]);
        async_b128(gb + 8, &lB[buf][trow][tcol + 8]);
#endif
    };
    auto wait_tiles = [&]() {
#if USE_TDM
        wait_tensor0();
#else
        wait_async0();
#endif
    };

    v8f acc[4][2] = {};
    prefetch(0, 0);
    for (int k0 = 0; k0 < HID; k0 += 32) {
        const int buf = (k0 >> 5) & 1;
        wait_tiles();
        __syncthreads();
        if (k0 + 32 < HID) prefetch(k0 + 32, buf ^ 1);

        v16bf a[4], b[2];
        #pragma unroll
        for (int i = 0; i < 4; ++i) a[i] = load_a_frag(&lA[buf][0][0], 32, Mwl + i * 16, 0);
        #pragma unroll
        for (int j = 0; j < 2; ++j) b[j] = load_b_frag(&lB[buf][0][0], 32, Nwl + j * 16, 0);
        #pragma unroll
        for (int i = 0; i < 4; ++i)
            #pragma unroll
            for (int j = 0; j < 2; ++j)
                acc[i][j] = wmma_bf16(a[i], b[j], acc[i][j]);
    }

    const int lane = threadIdx.x & 31, half = lane >> 4, nl = lane & 15;
    #pragma unroll
    for (int i = 0; i < 4; ++i)
        #pragma unroll
        for (int j = 0; j < 2; ++j)
            #pragma unroll
            for (int r = 0; r < 8; ++r)
                out[(size_t)(Mbase + Mwl + i * 16 + r + 8 * half) * HID +
                    Nbase + Nwl + j * 16 + nl] = acc[i][j][r];
}

// ---------------------------------------------------------------------------
extern "C" void kernel_launch(void* const* d_in, const int* in_sizes, int n_in,
                              void* d_out, int out_size, void* d_ws, size_t ws_size,
                              hipStream_t stream) {
    (void)in_sizes; (void)n_in; (void)out_size; (void)ws_size;

    const float* x  = (const float*)d_in[0];
    const float* Wq = (const float*)d_in[1];
    const float* Wk = (const float*)d_in[2];
    const float* Wv = (const float*)d_in[3];
    const float* Wo = (const float*)d_in[4];
    float* out = (float*)d_out;

    char* ws = (char*)d_ws;
    const size_t SZ_X = (size_t)BATCH * SEQ * HID * sizeof(__bf16);   // 16 MB
    const size_t SZ_W = (size_t)HID * HID * sizeof(__bf16);           //  8 MB
    __bf16* xbf  = (__bf16*)(ws);                         // reused for ctx later
    __bf16* wqb  = (__bf16*)(ws + SZ_X);
    __bf16* wkb  = (__bf16*)(ws + SZ_X + SZ_W);
    __bf16* wvb  = (__bf16*)(ws + SZ_X + 2 * SZ_W);
    __bf16* wob  = (__bf16*)(ws + SZ_X + 3 * SZ_W);
    __bf16* Qbf  = (__bf16*)(ws + SZ_X + 4 * SZ_W);
    __bf16* Kbf  = (__bf16*)(ws + 2 * SZ_X + 4 * SZ_W);
    __bf16* Vt   = (__bf16*)(ws + 3 * SZ_X + 4 * SZ_W);
    __bf16* ctxb = xbf;   // x is dead after the QKV GEMM

    const long long nx = (long long)BATCH * SEQ * HID;
    const long long nw = (long long)HID * HID;
    convert_bf16_kernel<<<4096, 256, 0, stream>>>(x,  xbf, nx);
    convert_bf16_kernel<<<2048, 256, 0, stream>>>(Wq, wqb, nw);
    convert_bf16_kernel<<<2048, 256, 0, stream>>>(Wk, wkb, nw);
    convert_bf16_kernel<<<2048, 256, 0, stream>>>(Wv, wvb, nw);
    convert_bf16_kernel<<<2048, 256, 0, stream>>>(Wo, wob, nw);

    qkv_gemm_kernel<<<dim3(HID / 128, (BATCH * SEQ) / 128, 3), 256, 0, stream>>>(
        xbf, wqb, wkb, wvb, Qbf, Kbf, Vt);

    rope_kernel<<<8192, 256, 0, stream>>>(Qbf, Kbf);

    attn_kernel<<<dim3(SEQ / 128, NH, BATCH), 256, 0, stream>>>(Qbf, Kbf, Vt, ctxb);

    out_gemm_kernel<<<dim3(HID / 128, (BATCH * SEQ) / 128), 256, 0, stream>>>(
        ctxb, wob, out);
}
